// GNRF_76647986365056
// MI455X (gfx1250) — matrix operand order repaired and checked by reading, not compiled, not run on hardware
//
#include <hip/hip_runtime.h>

// GNRF message-passing step for MI455X (gfx1250, wave32, WMMA).
// ~131 GFLOP of per-edge MLP GEMMs; node tables (~26 MB) are L2-resident
// (192 MB L2). All GEMM work goes through v_wmma_f32_16x16x32_f16.
// Round-2 changes (operand-bandwidth driven):
//  - M=32 edges per wave (two A register tiles) -> each B fragment feeds
//    two WMMAs, halving B traffic per FLOP and doubling WMMA density.
//  - Dominant [256,128] weight matrices staged in LDS per 256-thread block
//    (8 waves share one global fetch; B fragments become ds_load_b128).
// A fragments load as two contiguous 16B chunks per lane directly from
// gathered node rows (ISA 16-bit A 16x32 layout: k = 32*ks + 8g + [0,8), +16).
// B fragments (weights pre-transposed to f16 [n][k]) load like A, as two 16B
// chunks per lane: k = 32*ks + 8g + [0,8), +16.

#define D_HID 128

typedef __attribute__((ext_vector_type(16))) _Float16 v16h;
typedef __attribute__((ext_vector_type(8)))  _Float16 v8h;
typedef __attribute__((ext_vector_type(8)))  float    v8f;

union AFrag { v16h v; v8h h[2]; };

__device__ __forceinline__ v16h load_frag(const _Float16* p0, const _Float16* p1) {
  AFrag f;
  f.h[0] = *(const v8h*)p0;
  f.h[1] = *(const v8h*)p1;
  return f.v;
}

__device__ __forceinline__ float reduce16(float v) {
  v += __shfl_xor(v, 1, 32);
  v += __shfl_xor(v, 2, 32);
  v += __shfl_xor(v, 4, 32);
  v += __shfl_xor(v, 8, 32);
  return v;
}

#define ZERO8 {0.f,0.f,0.f,0.f,0.f,0.f,0.f,0.f}
#define WMMA(A, B, C) __builtin_amdgcn_wmma_f32_16x16x32_f16(false, (A), false, (B), (short)0, (C), false, false)

// ---------------------------------------------------------------- utilities
__global__ void zero_f32(float* __restrict__ p, long long n) {
  long long i = (long long)blockIdx.x * blockDim.x + threadIdx.x;
  long long stride = (long long)gridDim.x * blockDim.x;
  for (; i < n; i += stride) p[i] = 0.f;
}

__global__ void cvt_f32_to_f16(const float* __restrict__ in,
                               _Float16* __restrict__ out, long long n) {
  long long i = (long long)blockIdx.x * blockDim.x + threadIdx.x;
  long long stride = (long long)gridDim.x * blockDim.x;
  for (; i < n; i += stride) out[i] = (_Float16)in[i];
}

// W [K][Nc] f32 row-major -> WT [Nc][K] f16 (n-major, k-contiguous)
__global__ void transpose_to_f16(const float* __restrict__ W,
                                 _Float16* __restrict__ WT, int K, int Nc) {
  int idx = blockIdx.x * blockDim.x + threadIdx.x;
  if (idx >= K * Nc) return;
  int k = idx / Nc;
  int n = idx - k * Nc;
  WT[(long long)n * K + k] = (_Float16)W[idx];
}

// Row-normalize H -> Hn (f32) and Hn16 (f16). One wave per row.
__global__ __launch_bounds__(256) void normalize_rows(
    const float* __restrict__ H, float* __restrict__ Hn,
    _Float16* __restrict__ Hn16, int N) {
  int wid  = (int)((blockIdx.x * blockDim.x + threadIdx.x) >> 5);
  int lane = threadIdx.x & 31;
  if (wid >= N) return;
  const float* row = H + (long long)wid * D_HID;
  float4 v = *(const float4*)(row + lane * 4);
  float ss = v.x * v.x + v.y * v.y + v.z * v.z + v.w * v.w;
  ss = reduce16(ss);
  ss += __shfl_xor(ss, 16, 32);
  float inv = 1.0f / (sqrtf(ss) + 1e-8f);
  float4 o; o.x = v.x * inv; o.y = v.y * inv; o.z = v.z * inv; o.w = v.w * inv;
  *(float4*)(Hn + (long long)wid * D_HID + lane * 4) = o;
  _Float16* h16 = Hn16 + (long long)wid * D_HID + lane * 4;
  h16[0] = (_Float16)o.x; h16[1] = (_Float16)o.y;
  h16[2] = (_Float16)o.z; h16[3] = (_Float16)o.w;
}

__global__ void degree_count(const int* __restrict__ src,
                             float* __restrict__ cnt, int E) {
  int e = blockIdx.x * blockDim.x + threadIdx.x;
  if (e < E) atomicAdd(&cnt[src[e]], 1.0f);
}

// ------------------------------------------------------------------- pass 1
// Per 32-edge tile (one wave, two 16-row A tiles):
//   X = relu(concat(Hn16[src],Hn16[dst]) @ w1_1 + b1_1)  (B from LDS)
//   C = relu(X @ w1_2 + b1_2)                            (B from global/L1)
//   atomic scatter-add C into agg[src].
__global__ __launch_bounds__(256) void gnrf_pass1(
    const _Float16* __restrict__ Hn16,
    const int* __restrict__ src, const int* __restrict__ dst,
    const _Float16* __restrict__ w11T, const float* __restrict__ b11,
    const _Float16* __restrict__ w12T, const float* __restrict__ b12,
    float* __restrict__ agg, int E, int ntiles) {
  __shared__ __align__(16) _Float16 w11s[256 * 128];        // 64 KB
  __shared__ __align__(16) _Float16 Xtile[8][32][D_HID];    // 64 KB
  __shared__ int sIdxSm[8][32];

  // cooperative stage of w1_1^T into LDS (8 waves share it)
  {
    const uint4* s4 = (const uint4*)w11T;
    uint4* d4 = (uint4*)w11s;
    for (int i = threadIdx.x; i < 256 * 128 / 8; i += blockDim.x) d4[i] = s4[i];
  }
  __syncthreads();

  const int lane = threadIdx.x & 31;
  const int wave = threadIdx.x >> 5;
  const int g    = lane >> 4;
  const int ml   = lane & 15;
  const int tile = blockIdx.x * 8 + wave;
  if (tile >= ntiles) return;
  const int e0 = tile * 32;

  int eA = e0 + ml;      if (eA >= E) eA = E - 1;   // tail clamp (scatter guarded)
  int eB = e0 + 16 + ml; if (eB >= E) eB = E - 1;
  const int sA = src[eA], dA = dst[eA];
  const int sB = src[eB], dB = dst[eB];
  if (g == 0) { sIdxSm[wave][ml] = sA; sIdxSm[wave][16 + ml] = sB; }

  const _Float16* rsA = Hn16 + (long long)sA * D_HID;
  const _Float16* rdA = Hn16 + (long long)dA * D_HID;
  const _Float16* rsB = Hn16 + (long long)sB * D_HID;
  const _Float16* rdB = Hn16 + (long long)dB * D_HID;

  // A fragments, K=256 (8 k-steps of 32); ks<4 -> src half, ks>=4 -> dst half
  v16h af0[8], af1[8];
#pragma unroll
  for (int ks = 0; ks < 8; ++ks) {
    const int c0 = 32 * (ks & 3) + 8 * g;
    const _Float16* r0 = (ks < 4) ? rsA : rdA;
    const _Float16* r1 = (ks < 4) ? rsB : rdB;
    af0[ks] = load_frag(r0 + c0, r0 + c0 + 16);
    af1[ks] = load_frag(r1 + c0, r1 + c0 + 16);
  }

  // GEMM1: [32,256] x [256,128] -> relu -> Xtile f16 (B fragments from LDS)
#pragma unroll
  for (int nt = 0; nt < 8; ++nt) {
    v8f acc0 = ZERO8, acc1 = ZERO8;
    const _Float16* wrow = w11s + (16 * nt + ml) * 256 + 8 * g;
#pragma unroll
    for (int ks = 0; ks < 8; ++ks) {
      v16h b = load_frag(wrow + 32 * ks, wrow + 32 * ks + 16);
      acc0 = WMMA(af0[ks], b, acc0);
      acc1 = WMMA(af1[ks], b, acc1);
    }
    const int col = 16 * nt + ml;
    const float bias = b11[col];
#pragma unroll
    for (int r = 0; r < 8; ++r) {
      float v0 = acc0[r] + bias; v0 = v0 > 0.f ? v0 : 0.f;
      float v1 = acc1[r] + bias; v1 = v1 > 0.f ? v1 : 0.f;
      Xtile[wave][r + 8 * g][col]      = (_Float16)v0;  // C-frag (m=r+8g, n=col)
      Xtile[wave][16 + r + 8 * g][col] = (_Float16)v1;
    }
  }

  // A fragments for GEMM2 from LDS (K=128, 4 k-steps)
  v16h a20[4], a21[4];
#pragma unroll
  for (int ks = 0; ks < 4; ++ks) {
    const int c0 = 32 * ks + 8 * g;
    const _Float16* r0 = &Xtile[wave][ml][0];
    const _Float16* r1 = &Xtile[wave][16 + ml][0];
    a20[ks] = load_frag(r0 + c0, r0 + c0 + 16);
    a21[ks] = load_frag(r1 + c0, r1 + c0 + 16);
  }

  // GEMM2: [32,128] x [128,128] -> relu -> atomic scatter to agg[src]
#pragma unroll
  for (int nt = 0; nt < 8; ++nt) {
    v8f acc0 = ZERO8, acc1 = ZERO8;
    const _Float16* wrow = w12T + (16 * nt + ml) * 128 + 8 * g;
#pragma unroll
    for (int ks = 0; ks < 4; ++ks) {
      v16h b = load_frag(wrow + 32 * ks, wrow + 32 * ks + 16);
      acc0 = WMMA(a20[ks], b, acc0);
      acc1 = WMMA(a21[ks], b, acc1);
    }
    const int col = 16 * nt + ml;
    const float bias = b12[col];
#pragma unroll
    for (int r = 0; r < 8; ++r) {
      const int m0 = r + 8 * g;
      const int m1 = 16 + m0;
      if (e0 + m0 < E) {
        float v = acc0[r] + bias; v = v > 0.f ? v : 0.f;
        atomicAdd(&agg[(long long)sIdxSm[wave][m0] * D_HID + col], v);
      }
      if (e0 + m1 < E) {
        float v = acc1[r] + bias; v = v > 0.f ? v : 0.f;
        atomicAdd(&agg[(long long)sIdxSm[wave][m1] * D_HID + col], v);
      }
    }
  }
}

// ------------------------------------------------------------------- pass 2
// Per 32-edge tile: Y = relu(concat(agg16[src],agg16[dst]) @ w2_1 + b2_1),
// curv = Y @ w2_2 + b2_2 (per-row dot + cross-lane reduce),
// cos = <Hn[src],Hn[dst]>, msg = curv*(Hn[dst]-cos*Hn[src]) scattered to s[src].
__global__ __launch_bounds__(256) void gnrf_pass2(
    const float* __restrict__ Hn, const _Float16* __restrict__ agg16,
    const int* __restrict__ src, const int* __restrict__ dst,
    const _Float16* __restrict__ w21T, const float* __restrict__ b21,
    const float* __restrict__ w22, const float* __restrict__ b22,
    float* __restrict__ sbuf, int E, int ntiles) {
  __shared__ __align__(16) _Float16 w21s[256 * 128];   // 64 KB
  __shared__ int   sIdxSm[8][32];
  __shared__ int   dIdxSm[8][32];
  __shared__ float curvSm[8][32];
  __shared__ float cosSm[8][32];

  {
    const uint4* s4 = (const uint4*)w21T;
    uint4* d4 = (uint4*)w21s;
    for (int i = threadIdx.x; i < 256 * 128 / 8; i += blockDim.x) d4[i] = s4[i];
  }
  __syncthreads();

  const int lane = threadIdx.x & 31;
  const int wave = threadIdx.x >> 5;
  const int g    = lane >> 4;
  const int ml   = lane & 15;
  const int tile = blockIdx.x * 8 + wave;
  if (tile >= ntiles) return;
  const int e0 = tile * 32;

  int eA = e0 + ml;      if (eA >= E) eA = E - 1;
  int eB = e0 + 16 + ml; if (eB >= E) eB = E - 1;
  const int sA = src[eA], dA = dst[eA];
  const int sB = src[eB], dB = dst[eB];
  if (g == 0) {
    sIdxSm[wave][ml] = sA;      dIdxSm[wave][ml] = dA;
    sIdxSm[wave][16 + ml] = sB; dIdxSm[wave][16 + ml] = dB;
  }

  const _Float16* rsA = agg16 + (long long)sA * D_HID;
  const _Float16* rdA = agg16 + (long long)dA * D_HID;
  const _Float16* rsB = agg16 + (long long)sB * D_HID;
  const _Float16* rdB = agg16 + (long long)dB * D_HID;

  v16h af0[8], af1[8];
#pragma unroll
  for (int ks = 0; ks < 8; ++ks) {
    const int c0 = 32 * (ks & 3) + 8 * g;
    const _Float16* r0 = (ks < 4) ? rsA : rdA;
    const _Float16* r1 = (ks < 4) ? rsB : rdB;
    af0[ks] = load_frag(r0 + c0, r0 + c0 + 16);
    af1[ks] = load_frag(r1 + c0, r1 + c0 + 16);
  }

  // GEMM [32,256]x[256,128] (B from LDS), fused relu + per-row dot with w2_2
  float p0[8] = ZERO8, p1[8] = ZERO8;
#pragma unroll
  for (int nt = 0; nt < 8; ++nt) {
    v8f acc0 = ZERO8, acc1 = ZERO8;
    const _Float16* wrow = w21s + (16 * nt + ml) * 256 + 8 * g;
#pragma unroll
    for (int ks = 0; ks < 8; ++ks) {
      v16h b = load_frag(wrow + 32 * ks, wrow + 32 * ks + 16);
      acc0 = WMMA(af0[ks], b, acc0);
      acc1 = WMMA(af1[ks], b, acc1);
    }
    const int col = 16 * nt + ml;
    const float bias = b21[col];
    const float wv = w22[col];
#pragma unroll
    for (int r = 0; r < 8; ++r) {
      float v0 = acc0[r] + bias; v0 = v0 > 0.f ? v0 : 0.f;
      float v1 = acc1[r] + bias; v1 = v1 > 0.f ? v1 : 0.f;
      p0[r] += v0 * wv;           // row m = r+8g,    columns across lanes
      p1[r] += v1 * wv;           // row m = 16+r+8g
    }
  }
  const float bcurv = b22[0];
#pragma unroll
  for (int r = 0; r < 8; ++r) {
    p0[r] = reduce16(p0[r]) + bcurv;
    p1[r] = reduce16(p1[r]) + bcurv;
  }
  if (ml == 0) {
#pragma unroll
    for (int r = 0; r < 8; ++r) {
      curvSm[wave][r + 8 * g]      = p0[r];
      curvSm[wave][16 + r + 8 * g] = p1[r];
    }
  }

  // cosine per edge: lane covers k in [64g, 64g+64) for edges eA and eB
  {
    const float* hsA = Hn + (long long)sA * D_HID + 64 * g;
    const float* hdA = Hn + (long long)dA * D_HID + 64 * g;
    const float* hsB = Hn + (long long)sB * D_HID + 64 * g;
    const float* hdB = Hn + (long long)dB * D_HID + 64 * g;
    float c0 = 0.f, c1 = 0.f;
#pragma unroll
    for (int k = 0; k < 64; k += 4) {
      float4 a0 = *(const float4*)(hsA + k);
      float4 b0 = *(const float4*)(hdA + k);
      float4 a1 = *(const float4*)(hsB + k);
      float4 b1 = *(const float4*)(hdB + k);
      c0 += a0.x * b0.x + a0.y * b0.y + a0.z * b0.z + a0.w * b0.w;
      c1 += a1.x * b1.x + a1.y * b1.y + a1.z * b1.z + a1.w * b1.w;
    }
    c0 += __shfl_xor(c0, 16, 32);
    c1 += __shfl_xor(c1, 16, 32);
    if (g == 0) { cosSm[wave][ml] = c0; cosSm[wave][16 + ml] = c1; }
  }

  // damped edge message + scatter-add (32 edges x 128 dims over 32 lanes)
  for (int m = 0; m < 32; ++m) {
    if (e0 + m >= E) break;
    const int sn = sIdxSm[wave][m];
    const int dn = dIdxSm[wave][m];
    const float cv = curvSm[wave][m];
    const float cs = cosSm[wave][m];
    const float* hs = Hn + (long long)sn * D_HID;
    const float* hd = Hn + (long long)dn * D_HID;
#pragma unroll
    for (int j = 0; j < 4; ++j) {
      const int d = lane + 32 * j;
      const float v = cv * (hd[d] - cs * hs[d]);
      atomicAdd(&sbuf[(long long)sn * D_HID + d], v);
    }
  }
}

// ---------------------------------------------------------------- finalize
// Hm = s / max(cnt,1), then row-normalize. One wave per row.
__global__ __launch_bounds__(256) void finalize_rows(
    const float* __restrict__ sbuf, const float* __restrict__ cnt,
    float* __restrict__ out, int N) {
  int wid  = (int)((blockIdx.x * blockDim.x + threadIdx.x) >> 5);
  int lane = threadIdx.x & 31;
  if (wid >= N) return;
  const float c = fmaxf(cnt[wid], 1.0f);
  float4 v = *(const float4*)(sbuf + (long long)wid * D_HID + lane * 4);
  v.x /= c; v.y /= c; v.z /= c; v.w /= c;
  float ss = v.x * v.x + v.y * v.y + v.z * v.z + v.w * v.w;
  ss = reduce16(ss);
  ss += __shfl_xor(ss, 16, 32);
  const float inv = 1.0f / (sqrtf(ss) + 1e-8f);
  float4 o; o.x = v.x * inv; o.y = v.y * inv; o.z = v.z * inv; o.w = v.w * inv;
  *(float4*)(out + (long long)wid * D_HID + lane * 4) = o;
}

// ------------------------------------------------------------------- launch
extern "C" void kernel_launch(void* const* d_in, const int* in_sizes, int n_in,
                              void* d_out, int out_size, void* d_ws, size_t ws_size,
                              hipStream_t stream) {
  const float* H   = (const float*)d_in[1];
  const int*   ei  = (const int*)d_in[2];          // edge_index [2,E]
  const float* w11 = (const float*)d_in[3];
  const float* b11 = (const float*)d_in[4];
  const float* w12 = (const float*)d_in[5];
  const float* b12 = (const float*)d_in[6];
  const float* w21 = (const float*)d_in[7];
  const float* b21 = (const float*)d_in[8];
  const float* w22 = (const float*)d_in[9];
  const float* b22 = (const float*)d_in[10];

  const int N = in_sizes[1] / D_HID;
  const int E = in_sizes[2] / 2;
  const int* src = ei;
  const int* dst = ei + E;

  // workspace carve-out (~103 MB total)
  char* ws = (char*)d_ws;
  size_t off = 0;
  auto carve = [&](size_t bytes) -> void* {
    off = (off + 255) & ~(size_t)255;
    void* p = ws + off;
    off += bytes;
    return p;
  };
  float*    Hn    = (float*)   carve((size_t)N * D_HID * 4);
  _Float16* Hn16  = (_Float16*)carve((size_t)N * D_HID * 2);
  float*    agg   = (float*)   carve((size_t)N * D_HID * 4);
  _Float16* agg16 = (_Float16*)carve((size_t)N * D_HID * 2);
  float*    sbuf  = (float*)   carve((size_t)N * D_HID * 4);
  float*    cnt   = (float*)   carve((size_t)N * 4);
  _Float16* w11T  = (_Float16*)carve((size_t)256 * 128 * 2);
  _Float16* w12T  = (_Float16*)carve((size_t)128 * 128 * 2);
  _Float16* w21T  = (_Float16*)carve((size_t)256 * 128 * 2);
  (void)n_in; (void)out_size; (void)ws_size;

  const long long nd = (long long)N * D_HID;

  zero_f32<<<2048, 256, 0, stream>>>(agg,  nd);
  zero_f32<<<2048, 256, 0, stream>>>(sbuf, nd);
  zero_f32<<<1024, 256, 0, stream>>>(cnt,  (long long)N);

  transpose_to_f16<<<(256 * 128 + 255) / 256, 256, 0, stream>>>(w11, w11T, 256, 128);
  transpose_to_f16<<<(128 * 128 + 255) / 256, 256, 0, stream>>>(w12, w12T, 128, 128);
  transpose_to_f16<<<(256 * 128 + 255) / 256, 256, 0, stream>>>(w21, w21T, 256, 128);

  normalize_rows<<<(N * 32 + 255) / 256, 256, 0, stream>>>(H, Hn, Hn16, N);
  degree_count<<<(E + 255) / 256, 256, 0, stream>>>(src, cnt, E);

  const int ntiles = (E + 31) / 32;          // 32 edges per wave-tile
  const int pblocks = (ntiles + 7) / 8;      // 8 waves (tiles) per 256-thread block

  gnrf_pass1<<<pblocks, 256, 0, stream>>>(Hn16, src, dst, w11T, b11, w12T, b12,
                                          agg, E, ntiles);
  cvt_f32_to_f16<<<2048, 256, 0, stream>>>(agg, agg16, nd);
  gnrf_pass2<<<pblocks, 256, 0, stream>>>(Hn, agg16, src, dst, w21T, b21,
                                          w22, b22, sbuf, E, ntiles);
  finalize_rows<<<(N * 32 + 255) / 256, 256, 0, stream>>>(sbuf, cnt,
                                                          (float*)d_out, N);
}
